// MixerDiffAttention_24756191494230
// MI455X (gfx1250) — matrix operand order repaired and checked by reading, not compiled, not run on hardware
//
#include <hip/hip_runtime.h>
#include <stdint.h>

#define B_ 2
#define T_ 2048
#define DMODEL_ 2048
#define H_ 16
#define KVH_ 8
#define D_ 128
#define EPS_ 1e-6f
#define LAMBDA_INIT_ 0.7836057f   // 0.8 - 0.6*exp(-0.3*12)

typedef __attribute__((ext_vector_type(16))) __bf16 v16bf;
typedef __attribute__((ext_vector_type(8)))  float  v8f;
typedef __attribute__((ext_vector_type(4)))  int    v4i_;

union FragB16 { v16bf v; uint4 u[2]; };

#define ASG __attribute__((address_space(1)))
#define ASL __attribute__((address_space(3)))

#if defined(__has_builtin)
#if __has_builtin(__builtin_amdgcn_global_load_async_to_lds_b128)
#define HAVE_ASYNC_LDS 1
#endif
#endif

__device__ __forceinline__ unsigned short f2bf(float f) {
  union { float f; unsigned u; } c; c.f = f;
  unsigned r = c.u + 0x7FFFu + ((c.u >> 16) & 1u);  // round-to-nearest-even
  return (unsigned short)(r >> 16);
}

// ---------------------------------------------------------------------------
// GEMM: C[M,N] = A[M,K] * W[K,N], fp32 in/out, bf16 WMMA compute.
// Block tile 64x64, 8 waves (4 M x 2 N), k-step 32.
// ---------------------------------------------------------------------------
__global__ __launch_bounds__(256) void gemm_bf16_wmma(
    const float* __restrict__ A, const float* __restrict__ W,
    float* __restrict__ C, int M, int N, int K)
{
  __shared__ __align__(16) unsigned short at[64 * 32];   // A tile, row-major
  __shared__ __align__(16) unsigned short wt[64 * 32];   // W tile transposed [n][k]
  const int tid  = threadIdx.x;
  const int lane = tid & 31, wid = tid >> 5;
  const int lh = lane & 15, hi = lane >> 4;
  const int wm = wid >> 1, wn = wid & 1;
  const int row0 = blockIdx.x * 64;
  const int col0 = blockIdx.y * 64;

  v8f acc0 = {0,0,0,0,0,0,0,0};
  v8f acc1 = acc0;

  for (int k0 = 0; k0 < K; k0 += 32) {
    {   // load A tile 64x32 (fp32 -> bf16)
      int r = tid >> 2, c8 = (tid & 3) * 8;
      const float4* ap = (const float4*)&A[(size_t)(row0 + r) * K + k0 + c8];
      float4 a0 = ap[0], a1 = ap[1];
      unsigned short* d = &at[r * 32 + c8];
      d[0]=f2bf(a0.x); d[1]=f2bf(a0.y); d[2]=f2bf(a0.z); d[3]=f2bf(a0.w);
      d[4]=f2bf(a1.x); d[5]=f2bf(a1.y); d[6]=f2bf(a1.z); d[7]=f2bf(a1.w);
    }
    {   // load W tile 32x64, store transposed wt[n][k]
      int kk = tid >> 3, c8 = (tid & 7) * 8;
      const float4* wp = (const float4*)&W[(size_t)(k0 + kk) * N + col0 + c8];
      float4 w0 = wp[0], w1 = wp[1];
      float wv[8] = {w0.x,w0.y,w0.z,w0.w,w1.x,w1.y,w1.z,w1.w};
      #pragma unroll
      for (int q = 0; q < 8; ++q) wt[(c8 + q) * 32 + kk] = f2bf(wv[q]);
    }
    __syncthreads();

    FragB16 a, b0, b1;
    // A-frag (16-bit 16x32 layout): lanes 0-15 row=lane K{0..7,16..23}; lanes 16-31 K{8..15,24..31}
    a.u[0] = *(const uint4*)&at[(wm*16 + lh)*32 + hi*8];
    a.u[1] = *(const uint4*)&at[(wm*16 + lh)*32 + 16 + hi*8];
    int n0 = wn*32 + lh;
    b0.u[0] = *(const uint4*)&wt[n0*32 + hi*8];
    b0.u[1] = *(const uint4*)&wt[n0*32 + 16 + hi*8];
    int n1 = wn*32 + 16 + lh;
    b1.u[0] = *(const uint4*)&wt[n1*32 + hi*8];
    b1.u[1] = *(const uint4*)&wt[n1*32 + 16 + hi*8];

    acc0 = __builtin_amdgcn_wmma_f32_16x16x32_bf16(false, a.v, false, b0.v, (short)0, acc0, false, false);
    acc1 = __builtin_amdgcn_wmma_f32_16x16x32_bf16(false, a.v, false, b1.v, (short)0, acc1, false, false);
    __syncthreads();
  }

  #pragma unroll
  for (int r = 0; r < 8; ++r) {
    int row = row0 + wm*16 + r + hi*8;
    C[(size_t)row * N + col0 + wn*32 + lh]      = acc0[r];
    C[(size_t)row * N + col0 + wn*32 + 16 + lh] = acc1[r];
  }
}

// ---------------------------------------------------------------------------
// Q post: RMSnorm + RoPE + softmax_scaler*log(pos), fp32 -> bf16 head-major.
// One wave per (b,t,h): lane handles 4 of 128 dims.
// ---------------------------------------------------------------------------
__global__ __launch_bounds__(256) void post_q(
    const float* __restrict__ qf, const float* __restrict__ scaler,
    unsigned short* __restrict__ qb)
{
  int lane = threadIdx.x & 31, wid = threadIdx.x >> 5;
  int g = blockIdx.x * 8 + wid;     // row*16 + h
  int h = g & 15;
  int row = g >> 4;
  int b = row / T_;
  int t = row - b * T_;
  const float* src = &qf[(size_t)row * (H_*D_) + (size_t)h * D_ + lane*4];
  float4 x4 = *(const float4*)src;
  float xv[4] = {x4.x, x4.y, x4.z, x4.w};
  float ss = xv[0]*xv[0] + xv[1]*xv[1] + xv[2]*xv[2] + xv[3]*xv[3];
  #pragma unroll
  for (int m = 1; m < 32; m <<= 1) ss += __shfl_xor(ss, m, 32);
  float rms = rsqrtf(ss * (1.0f/128.0f) + EPS_);
  #pragma unroll
  for (int q = 0; q < 4; ++q) xv[q] *= rms;

  float sc = scaler[h] * __logf((float)(t + 1));
  float tt = (float)t;
  int dbase = lane * 4;
  float ov[4];
  #pragma unroll
  for (int q = 0; q < 4; ++q) {
    float px = __shfl_xor(xv[q], 16, 32);   // partner: d <-> d+64
    int d = dbase + q;
    int i = d & 63;
    float ang = tt * __powf(10000.0f, -(float)i * (1.0f/64.0f));
    float cs = __cosf(ang), sn = __sinf(ang);
    float o = (d < 64) ? (xv[q]*cs + px*sn) : (-px*sn + xv[q]*cs);
    ov[q] = o * sc;
  }
  size_t oidx = (((size_t)(b*H_ + h) * T_) + t) * D_ + dbase;
  unsigned int p0 = f2bf(ov[0]) | ((unsigned)f2bf(ov[1]) << 16);
  unsigned int p1 = f2bf(ov[2]) | ((unsigned)f2bf(ov[3]) << 16);
  *(uint2*)&qb[oidx] = make_uint2(p0, p1);
}

// K post: RMSnorm + RoPE, no scaling; 8 kv heads.
__global__ __launch_bounds__(256) void post_k(
    const float* __restrict__ kf, unsigned short* __restrict__ kb)
{
  int lane = threadIdx.x & 31, wid = threadIdx.x >> 5;
  int g = blockIdx.x * 8 + wid;     // row*8 + h
  int h = g & 7;
  int row = g >> 3;
  int b = row / T_;
  int t = row - b * T_;
  const float* src = &kf[(size_t)row * (KVH_*D_) + (size_t)h * D_ + lane*4];
  float4 x4 = *(const float4*)src;
  float xv[4] = {x4.x, x4.y, x4.z, x4.w};
  float ss = xv[0]*xv[0] + xv[1]*xv[1] + xv[2]*xv[2] + xv[3]*xv[3];
  #pragma unroll
  for (int m = 1; m < 32; m <<= 1) ss += __shfl_xor(ss, m, 32);
  float rms = rsqrtf(ss * (1.0f/128.0f) + EPS_);
  #pragma unroll
  for (int q = 0; q < 4; ++q) xv[q] *= rms;

  float tt = (float)t;
  int dbase = lane * 4;
  float ov[4];
  #pragma unroll
  for (int q = 0; q < 4; ++q) {
    float px = __shfl_xor(xv[q], 16, 32);
    int d = dbase + q;
    int i = d & 63;
    float ang = tt * __powf(10000.0f, -(float)i * (1.0f/64.0f));
    float cs = __cosf(ang), sn = __sinf(ang);
    ov[q] = (d < 64) ? (xv[q]*cs + px*sn) : (-px*sn + xv[q]*cs);
  }
  size_t oidx = (((size_t)(b*KVH_ + h) * T_) + t) * D_ + dbase;
  unsigned int p0 = f2bf(ov[0]) | ((unsigned)f2bf(ov[1]) << 16);
  unsigned int p1 = f2bf(ov[2]) | ((unsigned)f2bf(ov[3]) << 16);
  *(uint2*)&kb[oidx] = make_uint2(p0, p1);
}

// V post: fp32 -> bf16, relayout [b, vh, t, 256].
__global__ __launch_bounds__(256) void post_v(
    const float* __restrict__ vf, unsigned short* __restrict__ vb)
{
  int idx = blockIdx.x * 256 + threadIdx.x;
  size_t e = (size_t)idx * 4;
  int row = (int)(e >> 10);
  int c = (int)(e & 1023);
  int vh = c >> 8, cc = c & 255;
  int b = row / T_, t = row - b * T_;
  float4 v4 = *(const float4*)&vf[e];
  size_t o = (((size_t)(b*4 + vh) * T_) + t) * 256 + cc;
  unsigned int p0 = f2bf(v4.x) | ((unsigned)f2bf(v4.y) << 16);
  unsigned int p1 = f2bf(v4.z) | ((unsigned)f2bf(v4.w) << 16);
  *(uint2*)&vb[o] = make_uint2(p0, p1);
}

// ---------------------------------------------------------------------------
// Flash-style causal attention with WMMA + async global->LDS K staging.
// Block per (qtile32, vhead, b); 8 waves: mode x qhead x row-subtile.
// ---------------------------------------------------------------------------
__global__ __launch_bounds__(256) void attn_wmma(
    const unsigned short* __restrict__ qb, const unsigned short* __restrict__ kb,
    const unsigned short* __restrict__ vb, float* __restrict__ y1, float* __restrict__ y2)
{
  __shared__ __align__(16) unsigned short k1t[32 * 128];
  __shared__ __align__(16) unsigned short k2t[32 * 128];
  __shared__ __align__(16) unsigned short vtt[256 * 32];   // transposed [c][k]
  __shared__ __align__(16) unsigned short ps[8][16 * 32];  // per-wave P scratch

  const int tid  = threadIdx.x;
  const int lane = tid & 31, wid = tid >> 5;
  const int lh = lane & 15, hi = lane >> 4;
  const int qt = blockIdx.x;         // 0..63
  const int vh = blockIdx.y;         // 0..3
  const int b  = blockIdx.z;         // 0..1
  const int mode = wid & 1;
  const int j    = (wid >> 1) & 1;
  const int sub  = wid >> 2;
  const int hh = vh * 2 + j;         // attention head 0..7
  const int qhead = mode * 8 + hh;   // q1 = heads 0..7, q2 = heads 8..15
  const int qbase = qt * 32 + sub * 16;

  FragB16 qf[4];
  {
    const unsigned short* qr = &qb[(((size_t)(b*H_ + qhead) * T_) + qbase + lh) * D_];
    #pragma unroll
    for (int dc = 0; dc < 4; ++dc) {
      qf[dc].u[0] = *(const uint4*)&qr[dc*32 + hi*8];
      qf[dc].u[1] = *(const uint4*)&qr[dc*32 + 16 + hi*8];
    }
  }

  v8f zf = {0,0,0,0,0,0,0,0};
  v8f acc[16];
  #pragma unroll
  for (int nt = 0; nt < 16; ++nt) acc[nt] = zf;
  float mrow[8], lrow[8];
  #pragma unroll
  for (int r = 0; r < 8; ++r) { mrow[r] = -1e30f; lrow[r] = 0.f; }

  const int kh1 = vh, kh2 = 4 + vh;  // k1 = kv heads 0..3, k2 = kv heads 4..7
  const size_t kb1base = ((size_t)(b*KVH_ + kh1) * T_) * D_;
  const size_t kb2base = ((size_t)(b*KVH_ + kh2) * T_) * D_;
  const size_t vbbase  = ((size_t)(b*4 + vh) * T_) * 256;

  const int nsteps = qt + 1;
  for (int kt = 0; kt < nsteps; ++kt) {
    const int kbase = kt * 32;

    // --- stage K1/K2 tiles (32x128 bf16) via async global->LDS DMA ---
    for (int u = tid; u < 512; u += 256) {
      int r = u >> 4, c8 = (u & 15) * 8;
      const unsigned short* s1p = &kb[kb1base + (size_t)(kbase + r) * D_ + c8];
      const unsigned short* s2p = &kb[kb2base + (size_t)(kbase + r) * D_ + c8];
#ifdef HAVE_ASYNC_LDS
      __builtin_amdgcn_global_load_async_to_lds_b128(
          (ASG v4i_*)(size_t)s1p,
          (ASL v4i_*)(size_t)(unsigned)(size_t)&k1t[r*128 + c8], 0, 0);
      __builtin_amdgcn_global_load_async_to_lds_b128(
          (ASG v4i_*)(size_t)s2p,
          (ASL v4i_*)(size_t)(unsigned)(size_t)&k2t[r*128 + c8], 0, 0);
#else
      *(uint4*)&k1t[r*128 + c8] = *(const uint4*)s1p;
      *(uint4*)&k2t[r*128 + c8] = *(const uint4*)s2p;
#endif
    }
    // --- stage V tile 32x256 transposed (needs scatter, stays synchronous) ---
    for (int u = tid; u < 1024; u += 256) {
      int r = u >> 5, c8 = (u & 31) * 8;
      uint4 d4 = *(const uint4*)&vb[vbbase + (size_t)(kbase + r) * 256 + c8];
      const unsigned short* dd = (const unsigned short*)&d4;
      #pragma unroll
      for (int q = 0; q < 8; ++q) vtt[(c8 + q)*32 + r] = dd[q];
    }
    // prefetch next key-step K/V rows into cache while this step computes
    if (kt + 1 < nsteps) {
      int r = lane;
      __builtin_prefetch(&kb[kb1base + (size_t)(kbase + 32 + r) * D_], 0, 1);
      __builtin_prefetch(&kb[kb2base + (size_t)(kbase + 32 + r) * D_], 0, 1);
      __builtin_prefetch(&vb[vbbase + (size_t)(kbase + 32 + r) * 256], 0, 1);
    }
#ifdef HAVE_ASYNC_LDS
#if defined(__has_builtin) && __has_builtin(__builtin_amdgcn_s_wait_asynccnt)
    __builtin_amdgcn_s_wait_asynccnt(0);
#else
    asm volatile("s_wait_asynccnt 0" ::: "memory");
#endif
#endif
    __syncthreads();

    if (kbase <= qbase + 15) {
      const unsigned short* ktile = mode ? k2t : k1t;
      v8f s0 = zf, s1 = zf;
      #pragma unroll
      for (int dc = 0; dc < 4; ++dc) {   // S = Q . K^T over D=128 in 32-chunks
        FragB16 bk;
        bk.u[0] = *(const uint4*)&ktile[lh*128 + dc*32 + hi*8];
        bk.u[1] = *(const uint4*)&ktile[lh*128 + dc*32 + 16 + hi*8];
        s0 = __builtin_amdgcn_wmma_f32_16x16x32_bf16(false, qf[dc].v, false, bk.v, (short)0, s0, false, false);
        bk.u[0] = *(const uint4*)&ktile[(16+lh)*128 + dc*32 + hi*8];
        bk.u[1] = *(const uint4*)&ktile[(16+lh)*128 + dc*32 + 16 + hi*8];
        s1 = __builtin_amdgcn_wmma_f32_16x16x32_bf16(false, qf[dc].v, false, bk.v, (short)0, s1, false, false);
      }
      const float isq = 0.088388347648318447f;   // 1/sqrt(128)
      float alpha[8];
      #pragma unroll
      for (int r = 0; r < 8; ++r) {              // online softmax per row
        int rowg = qbase + r + hi*8;
        float v0 = s0[r]*isq; if (kbase + lh      > rowg) v0 = -1e30f;
        float v1 = s1[r]*isq; if (kbase + 16 + lh > rowg) v1 = -1e30f;
        float rm = fmaxf(v0, v1);
        rm = fmaxf(rm, __shfl_xor(rm, 1, 32));
        rm = fmaxf(rm, __shfl_xor(rm, 2, 32));
        rm = fmaxf(rm, __shfl_xor(rm, 4, 32));
        rm = fmaxf(rm, __shfl_xor(rm, 8, 32));
        float mn = fmaxf(mrow[r], rm);
        float al = __expf(mrow[r] - mn);
        float p0 = __expf(v0 - mn);
        float p1 = __expf(v1 - mn);
        float rs = p0 + p1;
        rs += __shfl_xor(rs, 1, 32);
        rs += __shfl_xor(rs, 2, 32);
        rs += __shfl_xor(rs, 4, 32);
        rs += __shfl_xor(rs, 8, 32);
        lrow[r] = lrow[r]*al + rs;
        mrow[r] = mn;
        alpha[r] = al;
        ps[wid][(r + hi*8)*32 + lh]      = f2bf(p0);   // transpose P via LDS
        ps[wid][(r + hi*8)*32 + 16 + lh] = f2bf(p1);
      }
      #pragma unroll
      for (int nt = 0; nt < 16; ++nt) {
        #pragma unroll
        for (int r = 0; r < 8; ++r) acc[nt][r] *= alpha[r];
      }
      asm volatile("s_wait_dscnt 0" ::: "memory");
      FragB16 pf;   // P as A-frag (16x32)
      pf.u[0] = *(const uint4*)&ps[wid][lh*32 + hi*8];
      pf.u[1] = *(const uint4*)&ps[wid][lh*32 + 16 + hi*8];
      #pragma unroll
      for (int nt = 0; nt < 16; ++nt) {          // acc += P . V (256-dim v)
        FragB16 bv;
        bv.u[0] = *(const uint4*)&vtt[(nt*16 + lh)*32 + hi*8];
        bv.u[1] = *(const uint4*)&vtt[(nt*16 + lh)*32 + 16 + hi*8];
        acc[nt] = __builtin_amdgcn_wmma_f32_16x16x32_bf16(false, pf.v, false, bv.v, (short)0, acc[nt], false, false);
      }
    }
    __syncthreads();
  }

  float* yout = mode ? y2 : y1;
  #pragma unroll
  for (int r = 0; r < 8; ++r) {
    float inv = 1.0f / lrow[r];
    int rowg = qbase + r + hi*8;
    size_t base = (((size_t)(b*8 + hh) * T_) + rowg) * 256;
    #pragma unroll
    for (int nt = 0; nt < 16; ++nt)
      yout[base + nt*16 + lh] = acc[nt][r] * inv;
  }
}

// ---------------------------------------------------------------------------
// y = rmsnorm(y1 - lam*y2) * gn_weight; one wave per (b,t,hh).
// ---------------------------------------------------------------------------
__global__ __launch_bounds__(256) void diff_merge(
    const float* __restrict__ y1, const float* __restrict__ y2,
    const float* __restrict__ lq1, const float* __restrict__ lk1,
    const float* __restrict__ lq2, const float* __restrict__ lk2,
    const float* __restrict__ gn, float* __restrict__ out)
{
  int lane = threadIdx.x & 31, wid = threadIdx.x >> 5;
  int g = blockIdx.x * 8 + wid;   // (b*T + t)*8 + hh
  int hh = g & 7;
  int row = g >> 3;
  int b = row / T_, t = row - b * T_;

  float s1 = lq1[lane*2]*lk1[lane*2] + lq1[lane*2+1]*lk1[lane*2+1];
  float s2 = lq2[lane*2]*lk2[lane*2] + lq2[lane*2+1]*lk2[lane*2+1];
  #pragma unroll
  for (int m = 1; m < 32; m <<= 1) { s1 += __shfl_xor(s1, m, 32); s2 += __shfl_xor(s2, m, 32); }
  float lam = __expf(s1) - __expf(s2) + LAMBDA_INIT_;

  size_t base = (((size_t)(b*8 + hh) * T_) + t) * 256 + lane*8;
  float yv[8]; float ss = 0.f;
  #pragma unroll
  for (int q = 0; q < 8; q += 4) {
    float4 a = *(const float4*)&y1[base + q];
    float4 c = *(const float4*)&y2[base + q];
    yv[q+0] = a.x - lam*c.x; yv[q+1] = a.y - lam*c.y;
    yv[q+2] = a.z - lam*c.z; yv[q+3] = a.w - lam*c.w;
    ss += yv[q+0]*yv[q+0] + yv[q+1]*yv[q+1] + yv[q+2]*yv[q+2] + yv[q+3]*yv[q+3];
  }
  #pragma unroll
  for (int m = 1; m < 32; m <<= 1) ss += __shfl_xor(ss, m, 32);
  float rms = rsqrtf(ss * (1.0f/256.0f) + EPS_);

  size_t ob = (((size_t)(b*T_ + t)) * 8 + hh) * 256 + lane*8;
  const float* gw = &gn[(size_t)hh * 256 + lane*8];
  #pragma unroll
  for (int q = 0; q < 8; ++q) out[ob + q] = yv[q] * rms * gw[q];
}

// ---------------------------------------------------------------------------
extern "C" void kernel_launch(void* const* d_in, const int* in_sizes, int n_in,
                              void* d_out, int out_size, void* d_ws, size_t ws_size,
                              hipStream_t stream) {
  const float* x   = (const float*)d_in[0];
  const float* Wq  = (const float*)d_in[1];
  const float* Wk  = (const float*)d_in[2];
  const float* Wv  = (const float*)d_in[3];
  const float* lq1 = (const float*)d_in[4];
  const float* lk1 = (const float*)d_in[5];
  const float* lq2 = (const float*)d_in[6];
  const float* lk2 = (const float*)d_in[7];
  const float* scl = (const float*)d_in[8];
  const float* gn  = (const float*)d_in[9];

  char* ws = (char*)d_ws;
  float*          qf32 = (float*)(ws);                       // 32 MB
  float*          kf32 = (float*)(ws + 33554432);            // 16 MB
  float*          vf32 = (float*)(ws + 50331648);            // 16 MB
  unsigned short* qbw  = (unsigned short*)(ws + 67108864);   // 16 MB
  unsigned short* kbw  = (unsigned short*)(ws + 83886080);   //  8 MB
  unsigned short* vbw  = (unsigned short*)(ws + 92274688);   //  8 MB
  float*          y1   = (float*)(ws);                       // alias dead qf32
  float*          y2   = (float*)(ws + 33554432);            // alias dead kf32+vf32

  gemm_bf16_wmma<<<dim3(64, 32), 256, 0, stream>>>(x, Wq, qf32, 4096, 2048, 2048);
  gemm_bf16_wmma<<<dim3(64, 16), 256, 0, stream>>>(x, Wk, kf32, 4096, 1024, 2048);
  gemm_bf16_wmma<<<dim3(64, 16), 256, 0, stream>>>(x, Wv, vf32, 4096, 1024, 2048);
  post_q<<<8192, 256, 0, stream>>>(qf32, scl, qbw);
  post_k<<<4096, 256, 0, stream>>>(kf32, kbw);
  post_v<<<4096, 256, 0, stream>>>(vf32, vbw);
  attn_wmma<<<dim3(64, 4, 2), 256, 0, stream>>>(qbw, kbw, vbw, y1, y2);
  diff_merge<<<4096, 256, 0, stream>>>(y1, y2, lq1, lk1, lq2, lk2, gn, (float*)d_out);
}
